// BitLinear_80848464380130
// MI455X (gfx1250) — matrix operand order repaired and checked by reading, not compiled, not run on hardware
//
#include <hip/hip_runtime.h>

// ---------------------------------------------------------------------------
// BitLinear forward for MI455X (gfx1250), wave32, WMMA f16->f32,
// triple-buffered async global->LDS pipeline (ASYNCcnt in-order semantics:
// issue next tile, then s_wait_asynccnt 4 completes exactly the current one).
// Uniform loop: final iteration issues a dummy prefetch so the body is
// branch-free and compute_tile is instantiated exactly once (no acc copies).
// out[m,n] = scale * sum_k tanh(4.5*x[m,k]) * tanh(4.5*clip(w[n,k]))
// scale = mean(|clip(w)|) / sqrt(K)
// ---------------------------------------------------------------------------

typedef _Float16 v16h __attribute__((ext_vector_type(16)));
typedef _Float16 v8h  __attribute__((ext_vector_type(8)));
typedef _Float16 v4h  __attribute__((ext_vector_type(4)));
typedef float    v8f  __attribute__((ext_vector_type(8)));
typedef float    v4f  __attribute__((ext_vector_type(4)));

#define M_DIM 8192
#define N_DIM 11008
#define K_DIM 4096

#define BM 128
#define BN 128
#define BK 32
#define LDSS 40                    // LDS row stride in halves (80B: 16B-aligned, bank-safe)
#define TILE_HALVES (BM * LDSS)    // 5120 halves per matrix tile
#define TILE_BYTES  (TILE_HALVES * 2)          // 10240 B
#define BUF_BYTES   (2 * TILE_BYTES)           // A+B per buffer = 20480 B
#define NBUF 3
#define SMEM_BYTES  (NBUF * BUF_BYTES)         // 61440 B (< 64KB)
#define NPART 2048

// ---------------------------------------------------------------------------
// x -> tanh(4.5 x) in f16 (vectorized x4, NT loads: read-once stream)
// ---------------------------------------------------------------------------
__global__ void cvt_x_kernel(const float* __restrict__ x,
                             _Float16* __restrict__ xh, int n4) {
  const int stride = gridDim.x * blockDim.x;
  for (int i = blockIdx.x * blockDim.x + threadIdx.x; i < n4; i += stride) {
    v4f v = __builtin_nontemporal_load(((const v4f*)x) + i);
    v4h o;
    o[0] = (_Float16)tanhf(4.5f * v[0]);
    o[1] = (_Float16)tanhf(4.5f * v[1]);
    o[2] = (_Float16)tanhf(4.5f * v[2]);
    o[3] = (_Float16)tanhf(4.5f * v[3]);
    ((v4h*)xh)[i] = o;
  }
}

// ---------------------------------------------------------------------------
// w -> tanh(4.5 clip(w)) in f16, plus per-block partial sums of |clip(w)|
// ---------------------------------------------------------------------------
__global__ void cvt_w_kernel(const float* __restrict__ w,
                             _Float16* __restrict__ wh,
                             float* __restrict__ partials, int n4) {
  __shared__ float red[256];
  const int stride = gridDim.x * blockDim.x;
  float s = 0.0f;
  for (int i = blockIdx.x * blockDim.x + threadIdx.x; i < n4; i += stride) {
    v4f v = __builtin_nontemporal_load(((const v4f*)w) + i);
    v4h o;
#pragma unroll
    for (int j = 0; j < 4; ++j) {
      float c = fminf(1.5f, fmaxf(-1.5f, v[j]));
      s += fabsf(c);
      o[j] = (_Float16)tanhf(4.5f * c);
    }
    ((v4h*)wh)[i] = o;
  }
  red[threadIdx.x] = s;
  __syncthreads();
#pragma unroll
  for (int off = 128; off > 0; off >>= 1) {
    if (threadIdx.x < (unsigned)off) red[threadIdx.x] += red[threadIdx.x + off];
    __syncthreads();
  }
  if (threadIdx.x == 0) partials[blockIdx.x] = red[0];
}

// ---------------------------------------------------------------------------
// reduce partials -> scale = (sum / (N*K)) / sqrt(K)     (deterministic tree)
// ---------------------------------------------------------------------------
__global__ void finalize_kernel(const float* __restrict__ partials,
                                float* __restrict__ scale) {
  __shared__ float red[256];
  float s = 0.0f;
  for (int i = threadIdx.x; i < NPART; i += 256) s += partials[i];
  red[threadIdx.x] = s;
  __syncthreads();
#pragma unroll
  for (int off = 128; off > 0; off >>= 1) {
    if (threadIdx.x < (unsigned)off) red[threadIdx.x] += red[threadIdx.x + off];
    __syncthreads();
  }
  if (threadIdx.x == 0) {
    float gamma = red[0] / (float)((long long)N_DIM * K_DIM);
    *scale = gamma / 64.0f;  // sqrt(4096) = 64
  }
}

// ---------------------------------------------------------------------------
// Async copy helpers (CDNA5 GLOBAL_LOAD_ASYNC_TO_LDS, tracked by ASYNCcnt)
// ---------------------------------------------------------------------------
__device__ __forceinline__ void async_b128(unsigned lds_byte_addr,
                                           const _Float16* gptr) {
  asm volatile("global_load_async_to_lds_b128 %0, %1, off"
               :
               : "v"(lds_byte_addr), "v"(gptr)
               : "memory");
}
__device__ __forceinline__ void wait_async4() {  // current tile done, next
  asm volatile("s_wait_asynccnt 0x4" ::: "memory");  // tile stays in flight
}

// ---------------------------------------------------------------------------
// WMMA GEMM: C[M,N] = scale * Xh[M,K] * Wh[N,K]^T
// 128x128 block tile, BK=32, 8 waves; each wave = 2x4 tiles of 16x16.
// Triple-buffered LDS fed by async global->LDS copies issued ahead of the
// wait so they cover wait + barrier + compute of the current tile.
// ---------------------------------------------------------------------------
__global__ __launch_bounds__(256) void gemm_kernel(
    const _Float16* __restrict__ xh, const _Float16* __restrict__ wh,
    const float* __restrict__ scale_p, float* __restrict__ out) {
  extern __shared__ _Float16 smem[];  // no static LDS -> base offset 0

  const int tid  = threadIdx.x;
  const int lane = tid & 31;
  const int wave = tid >> 5;    // 0..7
  const int wm   = wave & 3;    // 32-row slab within block
  const int wn   = wave >> 2;   // 64-col slab within block
  const int m0   = blockIdx.y * BM;
  const int n0   = blockIdx.x * BN;

  const int lrow = lane & 15;   // M for A-frag / N for B-frag / N for C-frag
  const int hsel = lane >> 4;   // half-wave select

  v8f acc[2][4] = {};

  // async copy indexing: per row 32 halves = 4 x b128; 256 thr -> 64 rows/pass
  const int grow = tid >> 2;          // 0..63
  const int gcol = (tid & 3) * 8;     // halves

  const _Float16* Abase = xh + (size_t)m0 * K_DIM;
  const _Float16* Bbase = wh + (size_t)n0 * K_DIM;

  // issue one full A+B tile (4 async b128 per thread) into buffer `buf`
  auto issue_tile = [&](int buf, int k0) {
    const unsigned base = (unsigned)(buf * BUF_BYTES);
#pragma unroll
    for (int p = 0; p < 2; ++p) {
      const int r = p * 64 + grow;
      const unsigned la = base + (unsigned)(r * LDSS + gcol) * 2;
      async_b128(la, &Abase[(size_t)r * K_DIM + k0 + gcol]);
      async_b128(la + TILE_BYTES, &Bbase[(size_t)r * K_DIM + k0 + gcol]);
    }
  };

  issue_tile(0, 0);  // prologue

  // Uniform branch-free hot loop. Buffer `nxt` was last read two iterations
  // ago (fenced by the previous barrier); in-order ASYNCcnt means
  // s_wait_asynccnt 4 retires exactly the current tile's 4 copies.
  // Final iteration issues a harmless dummy prefetch of the k=0 tile
  // (never read; drained by S_ENDPGM's implicit wait-idle).
  int cur = 0;
  for (int k0 = 0; k0 < K_DIM; k0 += BK) {
    const int nxt = (cur == NBUF - 1) ? 0 : cur + 1;
    const int knext = (k0 + BK) & (K_DIM - 1);  // wraps to 0 on last iter
    issue_tile(nxt, knext);
    wait_async4();
    __syncthreads();

    const _Float16* sA = smem + cur * (BUF_BYTES / 2);
    const _Float16* sB = sA + TILE_HALVES;

    // A fragments: 16x32 f16; lanes 0-15: K {0..7,16..23}, lanes 16-31: +8
    v16h afrag[2];
    const int akoff = hsel * 8;
#pragma unroll
    for (int mt = 0; mt < 2; ++mt) {
      const int row = wm * 32 + mt * 16 + lrow;
      union { v16h v; v8h h[2]; } u;
      u.h[0] = *(const v8h*)&sA[row * LDSS + akoff];
      u.h[1] = *(const v8h*)&sA[row * LDSS + akoff + 16];
      afrag[mt] = u.v;
    }
    // B fragments: 32x16 f16; lanes 0-15: K {0..7,16..23}, lanes 16-31: +8
    v16h bfrag[4];
    const int bkoff = hsel * 8;
#pragma unroll
    for (int nt = 0; nt < 4; ++nt) {
      const int row = wn * 64 + nt * 16 + lrow;
      union { v16h v; v8h h[2]; } u;
      u.h[0] = *(const v8h*)&sB[row * LDSS + bkoff];
      u.h[1] = *(const v8h*)&sB[row * LDSS + bkoff + 16];
      bfrag[nt] = u.v;
    }
#pragma unroll
    for (int mt = 0; mt < 2; ++mt)
#pragma unroll
      for (int nt = 0; nt < 4; ++nt)
        acc[mt][nt] = __builtin_amdgcn_wmma_f32_16x16x32_f16(
            false, afrag[mt], false, bfrag[nt], (short)0, acc[mt][nt],
            false, false);

    cur = nxt;
  }

  const float scale = *scale_p;
  // C/D layout: VGPR r -> M = r + 8*hsel, N = lrow
#pragma unroll
  for (int mt = 0; mt < 2; ++mt) {
#pragma unroll
    for (int nt = 0; nt < 4; ++nt) {
      const int n = n0 + wn * 64 + nt * 16 + lrow;
#pragma unroll
      for (int r = 0; r < 8; ++r) {
        const int m = m0 + wm * 32 + mt * 16 + r + hsel * 8;
        __builtin_nontemporal_store(acc[mt][nt][r] * scale,
                                    &out[(size_t)m * N_DIM + n]);
      }
    }
  }
}

// ---------------------------------------------------------------------------
extern "C" void kernel_launch(void* const* d_in, const int* in_sizes, int n_in,
                              void* d_out, int out_size, void* d_ws,
                              size_t ws_size, hipStream_t stream) {
  const float* x = (const float*)d_in[0];   // [4,2048,4096] f32
  const float* w = (const float*)d_in[1];   // [11008,4096] f32
  float* out = (float*)d_out;               // [4,2048,11008] f32

  const size_t xh_elems = (size_t)M_DIM * K_DIM;
  const size_t wh_elems = (size_t)N_DIM * K_DIM;

  char* ws = (char*)d_ws;
  _Float16* xh = (_Float16*)ws;
  size_t off = (xh_elems * 2 + 255) & ~(size_t)255;
  _Float16* wh = (_Float16*)(ws + off);
  off += (wh_elems * 2 + 255) & ~(size_t)255;
  float* partials = (float*)(ws + off);
  float* scale = partials + NPART;

  cvt_w_kernel<<<NPART, 256, 0, stream>>>(w, wh, partials, (int)(wh_elems / 4));
  cvt_x_kernel<<<2048, 256, 0, stream>>>(x, xh, (int)(xh_elems / 4));
  finalize_kernel<<<1, 256, 0, stream>>>(partials, scale);

  dim3 grid(N_DIM / BN, M_DIM / BM);  // 86 x 64
  gemm_kernel<<<grid, 256, SMEM_BYTES, stream>>>(xh, wh, scale, out);
}
